// Estimator_77764677861502
// MI455X (gfx1250) — matrix operand order repaired and checked by reading, not compiled, not run on hardware
//
#include <hip/hip_runtime.h>
#include <hip/hip_bf16.h>

typedef _Float16 h16;
typedef __attribute__((ext_vector_type(16))) _Float16 v16h;
typedef __attribute__((ext_vector_type(8)))  _Float16 v8h;
typedef __attribute__((ext_vector_type(8)))  float    v8f;
typedef __attribute__((ext_vector_type(4)))  unsigned int u32x4;
typedef __attribute__((ext_vector_type(8)))  int i32x8;
typedef __attribute__((ext_vector_type(4)))  int i32x4;

#define HH  256
#define WW_ 256
#define HWW 65536
#define BB  2

// ---------------------------------------------------------------------------
// Weight packing: swizzle OIHW f32 weights into per-lane-contiguous WMMA A
// fragments (16x32 f16, contiguous k = 16*(lane/16) + e), zero-padded in co and ci.
// Packed index = (((coTile*KS + kp)*nChunk + chunk)*32 + lane)*16 + e
// ---------------------------------------------------------------------------
__global__ void pack_w_kernel(const float* __restrict__ w, h16* __restrict__ wp,
                              int realCo, int realCi, int KS, int nChunk, int total)
{
    int idx = blockIdx.x * blockDim.x + threadIdx.x;
    if (idx >= total) return;
    int e    = idx & 15;
    int lane = (idx >> 4) & 31;
    int rest = idx >> 9;
    int chunk = rest % nChunk; rest /= nChunk;
    int kp    = rest % KS;     rest /= KS;
    int coTile = rest;
    int co = coTile * 16 + (lane & 15);
    int v = e >> 1, lo = e & 1;
    int base = (lane < 16) ? 0 : 16;
    int k  = base + e;
    int ci = chunk * 32 + k;
    float val = (co < realCo && ci < realCi) ? w[((size_t)co * realCi + ci) * KS + kp] : 0.0f;
    wp[idx] = (h16)val;
}

__global__ void pack_b_kernel(const float* __restrict__ b, float* __restrict__ bp,
                              int realCo, int coPad)
{
    int i = threadIdx.x;
    if (i < coPad) bp[i] = (i < realCo) ? b[i] : 0.0f;
}

// ---------------------------------------------------------------------------
// Fused correlation (9x9 window, 48ch, mean + leaky relu) + concat + f32->f16
// conversion into channel-tiled conv1 input (8 chunks of 32 channels).
// One block = 16x16 pixel tile; feat1 patch (24x24x48) staged in LDS as f16.
// ---------------------------------------------------------------------------
__global__ __launch_bounds__(256) void corr_pack_kernel(
    const float* __restrict__ feat0, const float* __restrict__ feat1,
    const float* __restrict__ last_feat, const float* __restrict__ last_flow,
    h16* __restrict__ act0)
{
    __shared__ h16 patch[24 * 24 * 48];
    const int tid = threadIdx.x;
    const int blk = blockIdx.x;
    const int bb  = blk >> 8;          // 256 tiles per batch
    const int t   = blk & 255;
    const int ty  = (t >> 4) << 4;
    const int tx  = (t & 15) << 4;

    for (int i = tid; i < 24 * 24; i += 256) {
        int py = i / 24, px = i % 24;
        int gy = ty + py - 4, gx = tx + px - 4;
        bool ok = (gy >= 0 && gy < HH && gx >= 0 && gx < WW_);
        #pragma unroll 4
        for (int c = 0; c < 48; ++c) {
            float v = ok ? feat1[(((size_t)bb * 48 + c) * HH + gy) * WW_ + gx] : 0.0f;
            patch[i * 48 + c] = (h16)v;
        }
    }
    __syncthreads();

    const int ly = tid >> 4, lx = tid & 15;
    const int y = ty + ly, x = tx + lx;
    const size_t pix = (size_t)y * WW_ + x;

    float x1[48];
    #pragma unroll 4
    for (int c = 0; c < 48; ++c)
        x1[c] = feat0[(((size_t)bb * 48 + c) * HH + y) * WW_ + x];

    auto st = [&](int ch, float v) {
        act0[(((size_t)(bb * 8 + (ch >> 5))) * HWW + pix) * 32 + (ch & 31)] = (h16)v;
    };

    for (int o = 0; o < 81; ++o) {
        int py = ly + o / 9, px = lx + o % 9;   // patch coords of (y+di, x+dj)
        const h16* pp = &patch[(py * 24 + px) * 48];
        float s = 0.0f;
        #pragma unroll 8
        for (int c = 0; c < 48; ++c) s += x1[c] * (float)pp[c];
        float v = s * (1.0f / 48.0f);
        st(o, v > 0.0f ? v : 0.1f * v);
    }
    for (int c = 0; c < 48; ++c) st(81 + c, x1[c]);
    for (int c = 0; c < 48; ++c)
        st(129 + c, feat1[(((size_t)bb * 48 + c) * HH + y) * WW_ + x]);
    for (int c = 0; c < 64; ++c)
        st(177 + c, last_feat[(((size_t)bb * 64 + c) * HH + y) * WW_ + x]);
    for (int c = 0; c < 4; ++c)
        st(241 + c, last_flow[(((size_t)bb * 4 + c) * HH + y) * WW_ + x]);
    for (int ch = 245; ch < 256; ++ch) st(ch, 0.0f);
}

// ---------------------------------------------------------------------------
// WMMA implicit-GEMM conv (KS = 1 or 9). Each wave: 32co x 32px tile,
// 4 f32 accumulators, K = nChunkIn*32 (*KS taps). Per-chunk weights are
// DMA'd into LDS by the Tensor Data Mover (one issue per block), A fragments
// then come from LDS; activations stream from global (L2-resident).
// ---------------------------------------------------------------------------
__device__ inline v8f lrelu8(v8f a) {
    #pragma unroll
    for (int r = 0; r < 8; ++r) { float v = a[r]; a[r] = v > 0.0f ? v : 0.1f * v; }
    return a;
}
__device__ inline v8h cvt8(v8f a) {
    v8h o;
    #pragma unroll
    for (int r = 0; r < 8; ++r) o[r] = (h16)a[r];
    return o;
}
__device__ inline void store_f32_frag(float* __restrict__ f32out, v8f f, int bb,
                                      int f32RealCo, int coBase, int t, int s,
                                      int py, int pxBase, int n, int lane)
{
    #pragma unroll
    for (int r = 0; r < 8; ++r) {
        int co = coBase + t * 16 + ((lane & 16) ? r + 8 : r);
        if (co < f32RealCo)
            f32out[((size_t)bb * f32RealCo + co) * HWW
                   + (size_t)py * WW_ + pxBase + s * 16 + n] = f[r];
    }
}

template <int KS>
__global__ __launch_bounds__(256) void conv_wmma_kernel(
    const h16* __restrict__ actIn, const h16* __restrict__ wp,
    const float* __restrict__ bp, h16* __restrict__ actOut,
    int nChunkIn, int nCoPair, float* __restrict__ f32out, int f32RealCo, int doLrelu)
{
    __shared__ h16 wlds[2 * KS * 512];        // 2 co-tiles x KS taps x 512 halfs

    const int lane   = threadIdx.x & 31;
    const int waveId = (blockIdx.x * blockDim.x + threadIdx.x) >> 5;
    const int pxTile = waveId & 8191;         // 8192 tiles = 4 x 2048 (B*HW/32 is 4096)
    const int coPair = waveId >> 13;          // uniform across the block
    const int bb     = pxTile >> 11;          // 2048 tiles per batch
    const int rem    = pxTile & 2047;
    const int py     = rem >> 3;              // row
    const int pxBase = (rem & 7) << 5;        // 8 tiles of 32 px per row
    const int n      = lane & 15;
    const int kHalf  = (lane & 16) ? 16 : 0;  // B-fragment channel half
    const int coBase = coPair * 32;
    const size_t HWl = HWW;

    v8f acc00, acc01, acc10, acc11;
    #pragma unroll
    for (int r = 0; r < 8; ++r) {
        int m = (lane & 16) ? r + 8 : r;
        float b0 = bp[coBase + m];
        float b1 = bp[coBase + 16 + m];
        acc00[r] = b0; acc01[r] = b0; acc10[r] = b1; acc11[r] = b1;
    }

    // TDM descriptor invariants for this block's weight tile:
    // 2*KS rows of 512 halfs (1KB), global row stride nChunkIn*1KB.
    const unsigned ldsOff = (unsigned)(size_t)(void*)&wlds[0];
    const unsigned long long gbase =
        (unsigned long long)(size_t)(wp + ((size_t)coPair * 2 * KS * nChunkIn) * 512);

    for (int chunk = 0; chunk < nChunkIn; ++chunk) {
        const h16* actC = actIn + ((size_t)bb * nChunkIn + chunk) * HWl * 32;
        if (chunk + 1 < nChunkIn)   // global_prefetch next chunk's activation row
            __builtin_prefetch(actC + HWl * 32 + ((size_t)py * WW_ + pxBase) * 32, 0, 0);

        __syncthreads();            // previous chunk's LDS weights no longer needed
        if (threadIdx.x < 32) {     // wave 0 drives the Tensor Data Mover
            unsigned long long ga = gbase + (unsigned long long)chunk * 1024ull;
            u32x4 g0;
            g0[0] = 1u;                                   // count=1 (valid user D#)
            g0[1] = ldsOff;                               // LDS byte address
            g0[2] = (unsigned)ga;                         // global addr [31:0]
            g0[3] = (unsigned)((ga >> 32) & 0x1FFFFFFu)   // global addr [56:32]
                    | (2u << 30);                         // type = 2 (image)
            i32x8 g1;
            g1[0] = 3 << 16;                              // data_size = 8 bytes
            g1[1] = 128 << 16;                            // tensor_dim0 = 128 elems
            g1[2] = (2 * KS) << 16;                       // tensor_dim1 = 2*KS rows
            g1[3] = 128 << 16;                            // tile_dim0 = 128 elems
            g1[4] = 2 * KS;                               // tile_dim1 = 2*KS rows
            g1[5] = nChunkIn * 128;                       // tensor_dim0_stride (elems)
            g1[6] = 0;
            g1[7] = 0;
            i32x4 z4 = {0, 0, 0, 0};
            i32x8 z8 = {0, 0, 0, 0, 0, 0, 0, 0};
            __builtin_amdgcn_tensor_load_to_lds(g0, g1, z4, z4, z8, 0);
            __builtin_amdgcn_s_wait_tensorcnt(0);
        }
        __syncthreads();            // LDS weights ready for all 8 waves

        #pragma unroll
        for (int kp = 0; kp < KS; ++kp) {
            const int ky = (KS == 9) ? kp / 3 - 1 : 0;
            const int kx = (KS == 9) ? kp % 3 - 1 : 0;
            v16h a0 = *(const v16h*)(wlds + (size_t)kp * 512 + lane * 16);
            v16h a1 = *(const v16h*)(wlds + (size_t)(KS + kp) * 512 + lane * 16);

            v16h bf0, bf1;
            #pragma unroll
            for (int i = 0; i < 16; ++i) { bf0[i] = (h16)0; bf1[i] = (h16)0; }
            const int yy = py + ky;
            if (yy >= 0 && yy < HH) {
                const int xx0 = pxBase + n + kx;
                const int xx1 = xx0 + 16;
                if (xx0 >= 0 && xx0 < WW_)
                    bf0 = *(const v16h*)(actC + ((size_t)yy * WW_ + xx0) * 32 + kHalf);
                if (xx1 >= 0 && xx1 < WW_)
                    bf1 = *(const v16h*)(actC + ((size_t)yy * WW_ + xx1) * 32 + kHalf);
            }
            acc00 = __builtin_amdgcn_wmma_f32_16x16x32_f16(false, a0, false, bf0, (short)0, acc00, false, false);
            acc10 = __builtin_amdgcn_wmma_f32_16x16x32_f16(false, a1, false, bf0, (short)0, acc10, false, false);
            acc01 = __builtin_amdgcn_wmma_f32_16x16x32_f16(false, a0, false, bf1, (short)0, acc01, false, false);
            acc11 = __builtin_amdgcn_wmma_f32_16x16x32_f16(false, a1, false, bf1, (short)0, acc11, false, false);
        }
    }

    if (doLrelu) {
        acc00 = lrelu8(acc00); acc01 = lrelu8(acc01);
        acc10 = lrelu8(acc10); acc11 = lrelu8(acc11);
    }

    if (actOut) {
        const size_t pbase = ((size_t)bb * nCoPair + coPair) * HWl + (size_t)py * WW_ + pxBase;
        const int hOff = (lane & 16) ? 8 : 0;
        *(v8h*)(actOut + (pbase + n)      * 32 +      hOff) = cvt8(acc00);
        *(v8h*)(actOut + (pbase + 16 + n) * 32 +      hOff) = cvt8(acc01);
        *(v8h*)(actOut + (pbase + n)      * 32 + 16 + hOff) = cvt8(acc10);
        *(v8h*)(actOut + (pbase + 16 + n) * 32 + 16 + hOff) = cvt8(acc11);
    }

    if (f32out) {
        store_f32_frag(f32out, acc00, bb, f32RealCo, coBase, 0, 0, py, pxBase, n, lane);
        store_f32_frag(f32out, acc01, bb, f32RealCo, coBase, 0, 1, py, pxBase, n, lane);
        store_f32_frag(f32out, acc10, bb, f32RealCo, coBase, 1, 0, py, pxBase, n, lane);
        store_f32_frag(f32out, acc11, bb, f32RealCo, coBase, 1, 1, py, pxBase, n, lane);
    }
}

// ---------------------------------------------------------------------------
extern "C" void kernel_launch(void* const* d_in, const int* in_sizes, int n_in,
                              void* d_out, int out_size, void* d_ws, size_t ws_size,
                              hipStream_t stream)
{
    (void)in_sizes; (void)n_in; (void)out_size; (void)ws_size;
    const float* feat0     = (const float*)d_in[0];
    const float* feat1     = (const float*)d_in[1];
    const float* last_feat = (const float*)d_in[2];
    const float* last_flow = (const float*)d_in[3];
    const float* W[6]; const float* Bv[6];
    for (int i = 0; i < 6; ++i) { W[i] = (const float*)d_in[4 + 2 * i]; Bv[i] = (const float*)d_in[5 + 2 * i]; }

    // layer table: realCo, realCi, KS, nChunkIn, coPad
    const int realCo[6]  = {160, 128, 112, 96, 64, 4};
    const int realCi[6]  = {245, 160, 128, 112, 96, 64};
    const int KSs[6]     = {1, 9, 9, 9, 9, 9};
    const int nChunk[6]  = {8, 5, 4, 4, 3, 2};
    const int coPad[6]   = {160, 128, 128, 96, 64, 32};

    char* ws = (char*)d_ws;
    size_t off = 0;
    auto alloc = [&](size_t bytes) -> char* {
        off = (off + 255) & ~(size_t)255;
        char* p = ws + off; off += bytes; return p;
    };

    h16*   wp[6]; float* bp[6]; int wtot[6];
    for (int i = 0; i < 6; ++i) {
        wtot[i] = (coPad[i] / 16) * KSs[i] * nChunk[i] * 512;
        wp[i] = (h16*)alloc((size_t)wtot[i] * sizeof(h16));
        bp[i] = (float*)alloc((size_t)coPad[i] * sizeof(float));
    }
    h16* bufA = (h16*)alloc((size_t)8 * BB * HWW * 32 * sizeof(h16)); // 64 MiB
    h16* bufB = (h16*)alloc((size_t)5 * BB * HWW * 32 * sizeof(h16)); // 40 MiB

    for (int i = 0; i < 6; ++i) {
        pack_w_kernel<<<(wtot[i] + 255) / 256, 256, 0, stream>>>(
            W[i], wp[i], realCo[i], realCi[i], KSs[i], nChunk[i], wtot[i]);
        pack_b_kernel<<<1, 256, 0, stream>>>(Bv[i], bp[i], realCo[i], coPad[i]);
    }

    corr_pack_kernel<<<BB * 256, 256, 0, stream>>>(feat0, feat1, last_feat, last_flow, bufA);

    float* out  = (float*)d_out;
    float* flow = out;                                   // (B,4,H,W)
    float* xout = out + (size_t)BB * 4 * HWW;            // (B,64,H,W)

    // grid blocks = coPadOut/32 * 8192 waves / 8 waves-per-block
    conv_wmma_kernel<1><<<5120, 256, 0, stream>>>(bufA, wp[0], bp[0], bufB, 8, 5, nullptr, 0, 1);
    conv_wmma_kernel<9><<<4096, 256, 0, stream>>>(bufB, wp[1], bp[1], bufA, 5, 4, nullptr, 0, 1);
    conv_wmma_kernel<9><<<4096, 256, 0, stream>>>(bufA, wp[2], bp[2], bufB, 4, 4, nullptr, 0, 1);
    conv_wmma_kernel<9><<<3072, 256, 0, stream>>>(bufB, wp[3], bp[3], bufA, 4, 3, nullptr, 0, 1);
    conv_wmma_kernel<9><<<2048, 256, 0, stream>>>(bufA, wp[4], bp[4], bufB, 3, 2, xout, 64, 1);
    conv_wmma_kernel<9><<<1024, 256, 0, stream>>>(bufB, wp[5], bp[5], nullptr, 2, 1, flow, 4, 0);
}
